// GAT_3040836846099
// MI455X (gfx1250) — compile-verified, matrix operand order checked by reading, not run on hardware
//
#include <hip/hip_runtime.h>
#include <hip/hip_bf16.h>

typedef unsigned short u16;
typedef __attribute__((ext_vector_type(16))) __bf16 v16bf;
typedef __attribute__((ext_vector_type(8)))  float  v8f;
typedef __attribute__((ext_vector_type(4)))  unsigned int v4u;
typedef __attribute__((ext_vector_type(8)))  int v8i;
typedef __attribute__((ext_vector_type(4)))  int v4i;

#define NODES 20000
#define EDGES 320000
#define GR    64
#define DD    128

#if __has_builtin(__builtin_amdgcn_tensor_load_to_lds) && __has_builtin(__builtin_amdgcn_s_wait_tensorcnt)
#define USE_TDM 1
#endif

// ---------------- workspace layout (bytes) ----------------
static constexpr size_t SZ_X   = (size_t)NODES * 768 * 4;          // 61.44 MB
static constexpr size_t OFF_X0 = 0;
static constexpr size_t OFF_X1 = OFF_X0 + SZ_X;
static constexpr size_t OFF_FT = OFF_X1 + SZ_X;
static constexpr size_t OFF_ABF = OFF_FT + SZ_X;                   // bf16 activations (<= N*512)
static constexpr size_t OFF_WBF = OFF_ABF + (size_t)NODES * 512 * 2;
static constexpr size_t OFF_EL  = OFF_WBF + (size_t)768 * 512 * 2;
static constexpr size_t OFF_ER  = OFF_EL + (size_t)NODES * 6 * 4;
static constexpr size_t OFF_M   = OFF_ER + (size_t)NODES * 6 * 4;
static constexpr size_t OFF_DEN = OFF_M  + (size_t)NODES * 6 * 4;
static constexpr size_t OFF_EB  = OFF_DEN + (size_t)NODES * 6 * 4;  // edge logits -> softmax numerators
static constexpr size_t OFF_PSUM = OFF_EB + (size_t)EDGES * 6 * 4;
static constexpr size_t OFF_PCNT = OFF_PSUM + (size_t)GR * DD * 4;

// ---------------- helpers ----------------
__device__ __forceinline__ u16 f2bf(float f) {
    unsigned u = __float_as_uint(f);
    unsigned r = (u + 0x7FFFu + ((u >> 16) & 1u)) >> 16;   // RNE
    return (u16)r;
}
__device__ __forceinline__ int   encf(float x) { int i = __float_as_int(x); return i >= 0 ? i : (i ^ 0x7FFFFFFF); }
__device__ __forceinline__ float decf(int i)   { return __int_as_float(i >= 0 ? i : (i ^ 0x7FFFFFFF)); }

// ---------------- fp32 -> bf16 conversion ----------------
__global__ void k_f32_to_bf16(const float* __restrict__ in, u16* __restrict__ out, long n) {
    long i = (long)blockIdx.x * blockDim.x + threadIdx.x;
    if (i < n) out[i] = f2bf(in[i]);
}

// W[K,N] fp32 -> Wt[N,K] bf16
__global__ void k_transpose_bf16(const float* __restrict__ W, u16* __restrict__ Wt, int K, int N) {
    long i = (long)blockIdx.x * blockDim.x + threadIdx.x;
    if (i < (long)K * N) {
        int n = (int)(i / K), k = (int)(i % K);
        Wt[i] = f2bf(W[(size_t)k * N + n]);
    }
}

// ---------------- WMMA bf16 GEMM: C[M,N] = A[M,K] * Bt[N,K]^T ----------------
// block tile 256(M) x 64(N); 8 waves, wave w -> rows [w*32, w*32+32) (two 16-row
// M-tiles) x 4 N-tiles => 8 WMMAs per 32-wide K step. B staged in 64x128 K-panels
// via the Tensor Data Mover (TDM); A streamed 256x32 per K step.
union Frag { uint4 q[2]; v16bf v; };

__global__ __launch_bounds__(256)
void gemm_bf16_wmma(const u16* __restrict__ A, const u16* __restrict__ Bt,
                    float* __restrict__ C, int M, int N, int K) {
    __shared__ __align__(16) u16 shA[256 * 32];    // 16 KB
    __shared__ __align__(16) u16 shB[64 * 128];    // 16 KB (one K-panel of B)

    const int tid  = threadIdx.x;
    const int lane = tid & 31;
    const int w    = tid >> 5;
    const int m15  = lane & 15;
    const int sel  = lane >> 4;                // K half select (ISA 16-bit A layout)
    const int mBase = blockIdx.y * 256;
    const int nBase = blockIdx.x * 64;
    const bool v0 = (mBase + w * 32) < M;       // wave-uniform tile validity (M%16==0)
    const bool v1 = (mBase + w * 32 + 16) < M;

    v8f acc0[4], acc1[4];
#pragma unroll
    for (int t = 0; t < 4; ++t)
#pragma unroll
        for (int j = 0; j < 8; ++j) { acc0[t][j] = 0.0f; acc1[t][j] = 0.0f; }

    for (int kp = 0; kp < K; kp += 128) {
        // ---- stage B panel: Wt rows [nBase, nBase+64) x cols [kp, kp+128) ----
#ifdef USE_TDM
        if (tid < 32) {                         // one wave drives the TDM
            unsigned long long ga = (unsigned long long)(size_t)&Bt[(size_t)nBase * K + kp];
            unsigned lds = (unsigned)(size_t)&shB[0];   // LDS aperture: low 32 bits = LDS addr
            v4u g0 = { 1u,                                   // count=1 valid descriptor
                       lds,                                  // lds_addr
                       (unsigned)ga,                         // global_addr[31:0]
                       (unsigned)((ga >> 32) & 0x01FFFFFFu) | 0x80000000u }; // ga[56:32] | type=2
            unsigned dim0 = (unsigned)K, dim1 = (unsigned)N;
            v8i g1 = { (int)0x00010000u,                     // data_size=1 (2 bytes)
                       (int)((dim0 & 0xFFFFu) << 16),        // tensor_dim0 lo16
                       (int)(((dim0 >> 16) & 0xFFFFu) | ((dim1 & 0xFFFFu) << 16)),
                       (int)(((dim1 >> 16) & 0xFFFFu) | (128u << 16)),   // tile_dim0 = 128
                       (int)64,                              // tile_dim1 = 64
                       (int)dim0,                            // tensor_dim0_stride lo32 = K
                       0, 0 };
            v4i gz4 = { 0, 0, 0, 0 };
            v8i gz8 = { 0, 0, 0, 0, 0, 0, 0, 0 };
            __builtin_amdgcn_tensor_load_to_lds(g0, g1, gz4, gz4, gz8, 0);
            __builtin_amdgcn_s_wait_tensorcnt(0);
        }
#else
        for (int s = tid; s < 64 * 128 / 8; s += 256) {
            int row = s >> 4, co = (s & 15) * 8;
            *(uint4*)&shB[row * 128 + co] =
                *(const uint4*)&Bt[(size_t)(nBase + row) * K + kp + co];
        }
#endif
        __syncthreads();

        for (int kb2 = 0; kb2 < 128; kb2 += 32) {
            const int kb = kp + kb2;
            // ---- stage A: 256 rows x 32 bf16 (4 x uint4 per thread), zero-fill OOB ----
#pragma unroll
            for (int ss = 0; ss < 4; ++ss) {
                int s   = tid + ss * 256;
                int row = s >> 2;
                int co  = (s & 3) * 8;
                int gRow = mBase + row;
                uint4 val = make_uint4(0u, 0u, 0u, 0u);
                if (gRow < M) {
                    val = *(const uint4*)&A[(size_t)gRow * K + kb + co];
                    if (kb + 32 < K)
                        __builtin_prefetch(&A[(size_t)gRow * K + kb + 32 + co], 0, 1);
                }
                *(uint4*)&shA[row * 32 + co] = val;
            }
            __syncthreads();

            // ---- fragments: load everything first, then 8 back-to-back WMMAs ----
            Frag a0, a1, b[4];
            a0.q[0] = *(const uint4*)&shA[(w * 32 + m15) * 32 + sel * 8];
            a0.q[1] = *(const uint4*)&shA[(w * 32 + m15) * 32 + 16 + sel * 8];
            a1.q[0] = *(const uint4*)&shA[(w * 32 + 16 + m15) * 32 + sel * 8];
            a1.q[1] = *(const uint4*)&shA[(w * 32 + 16 + m15) * 32 + 16 + sel * 8];
#pragma unroll
            for (int nt = 0; nt < 4; ++nt) {
                b[nt].q[0] = *(const uint4*)&shB[(nt * 16 + m15) * 128 + kb2 + sel * 8];
                b[nt].q[1] = *(const uint4*)&shB[(nt * 16 + m15) * 128 + kb2 + 16 + sel * 8];
            }
            if (v0) {
#pragma unroll
                for (int nt = 0; nt < 4; ++nt)
                    acc0[nt] = __builtin_amdgcn_wmma_f32_16x16x32_bf16(
                        false, a0.v, false, b[nt].v, (short)0, acc0[nt], false, false);
            }
            if (v1) {
#pragma unroll
                for (int nt = 0; nt < 4; ++nt)
                    acc1[nt] = __builtin_amdgcn_wmma_f32_16x16x32_bf16(
                        false, a1.v, false, b[nt].v, (short)0, acc1[nt], false, false);
            }
            __syncthreads();
        }
    }

    // ---- store C (per ISA C/D layout: row = v + 8*sel, col = lane&15) ----
    if (v0) {
#pragma unroll
        for (int nt = 0; nt < 4; ++nt) {
            float* cp = C + (size_t)(mBase + w * 32 + sel * 8) * N + nBase + nt * 16 + m15;
#pragma unroll
            for (int v = 0; v < 8; ++v) cp[(size_t)v * N] = acc0[nt][v];
        }
    }
    if (v1) {
#pragma unroll
        for (int nt = 0; nt < 4; ++nt) {
            float* cp = C + (size_t)(mBase + w * 32 + 16 + sel * 8) * N + nBase + nt * 16 + m15;
#pragma unroll
            for (int v = 0; v < 8; ++v) cp[(size_t)v * N] = acc1[nt][v];
        }
    }
}

// ---------------- attention logits: el/er[n,h] = feat[n,h,:] . al/ar[h,:] ----------------
__global__ __launch_bounds__(256)
void k_attn_logits(const float* __restrict__ feat, const float* __restrict__ al,
                   const float* __restrict__ ar, float* __restrict__ el,
                   float* __restrict__ er, int H) {
    int idx  = blockIdx.x * 8 + (threadIdx.x >> 5);   // one wave per (n,h)
    int lane = threadIdx.x & 31;
    if (idx >= NODES * H) return;
    int n = idx / H, h = idx % H;
    const float* f  = feat + ((size_t)n * H + h) * DD;
    const float* a1 = al + (size_t)h * DD;
    const float* a2 = ar + (size_t)h * DD;
    float sl = 0.f, sr = 0.f;
    for (int d = lane; d < DD; d += 32) { float v = f[d]; sl += v * a1[d]; sr += v * a2[d]; }
    for (int o = 16; o > 0; o >>= 1) { sl += __shfl_xor(sl, o, 32); sr += __shfl_xor(sr, o, 32); }
    if (lane == 0) { el[idx] = sl; er[idx] = sr; }
}

__global__ void k_init_m(int* __restrict__ m, long n) {
    long i = (long)blockIdx.x * blockDim.x + threadIdx.x;
    if (i < n) m[i] = encf(-1e30f);
}

__global__ void k_edge_logit_max(const float* __restrict__ el, const float* __restrict__ er,
                                 const int* __restrict__ src, const int* __restrict__ dst,
                                 float* __restrict__ ebuf, int* __restrict__ mbuf, int H) {
    long t = (long)blockIdx.x * blockDim.x + threadIdx.x;
    if (t >= (long)EDGES * H) return;
    int h = (int)(t % H), e = (int)(t / H);
    float x = el[(size_t)src[e] * H + h] + er[(size_t)dst[e] * H + h];
    x = x > 0.f ? x : 0.2f * x;
    ebuf[t] = x;
    atomicMax(&mbuf[(size_t)dst[e] * H + h], encf(x));
}

__global__ void k_edge_exp(const int* __restrict__ dst, float* __restrict__ ebuf,
                           const int* __restrict__ mbuf, float* __restrict__ den, int H) {
    long t = (long)blockIdx.x * blockDim.x + threadIdx.x;
    if (t >= (long)EDGES * H) return;
    int h = (int)(t % H), e = (int)(t / H);
    int di = dst[e];
    float m = decf(mbuf[(size_t)di * H + h]);
    if (m < -1e29f) m = 0.f;
    float a = __expf(ebuf[t] - m);
    ebuf[t] = a;
    atomicAdd(&den[(size_t)di * H + h], a);
}

__global__ void k_scatter(const int* __restrict__ src, const int* __restrict__ dst,
                          const float* __restrict__ ebuf, const float* __restrict__ den,
                          const float* __restrict__ feat, float* __restrict__ out, int H) {
    long t = (long)blockIdx.x * blockDim.x + threadIdx.x;
    long total = (long)EDGES * H * (DD / 4);
    if (t >= total) return;
    int c = (int)(t % (DD / 4));
    long r = t / (DD / 4);
    int h = (int)(r % H), e = (int)(r / H);
    int si = src[e], di = dst[e];
    float an = ebuf[(size_t)e * H + h] / fmaxf(den[(size_t)di * H + h], 1e-9f);
    const float4 f = *(const float4*)&feat[((size_t)si * H + h) * DD + c * 4];
    float* o = &out[((size_t)di * H + h) * DD + c * 4];
    atomicAdd(o + 0, an * f.x);
    atomicAdd(o + 1, an * f.y);
    atomicAdd(o + 2, an * f.z);
    atomicAdd(o + 3, an * f.w);
}

__global__ void k_bias_act(float* __restrict__ x, const float* __restrict__ b, int HD, int act) {
    long t = (long)blockIdx.x * blockDim.x + threadIdx.x;
    if (t >= (long)NODES * HD) return;
    int i = (int)(t % HD);
    float v = x[t] + b[i];
    if (act) v = v > 0.f ? v : (__expf(v) - 1.f);
    x[t] = v;
}

__global__ void k_mean_pool(const float* __restrict__ x3, const int* __restrict__ gid,
                            float* __restrict__ outLocal, float* __restrict__ psum, int H) {
    long t = (long)blockIdx.x * blockDim.x + threadIdx.x;
    if (t >= (long)NODES * DD) return;
    int n = (int)(t / DD), d = (int)(t % DD);
    float s = 0.f;
    for (int h = 0; h < H; ++h) s += x3[((size_t)n * H + h) * DD + d];
    s *= (1.0f / H);
    outLocal[t] = s;
    atomicAdd(&psum[(size_t)gid[n] * DD + d], s);
}

__global__ void k_counts(const int* __restrict__ gid, float* __restrict__ pcnt) {
    int n = blockIdx.x * blockDim.x + threadIdx.x;
    if (n < NODES) atomicAdd(&pcnt[gid[n]], 1.0f);
}

__global__ void k_final(const float* __restrict__ psum, const float* __restrict__ pcnt,
                        const float* __restrict__ Wm, const float* __restrict__ bm,
                        float* __restrict__ outG) {
    int t = blockIdx.x * blockDim.x + threadIdx.x;
    if (t >= GR * DD) return;
    int g = t / DD, j = t % DD;
    float inv = 1.0f / fmaxf(pcnt[g], 1.0f);
    float acc = bm[j];
    for (int k = 0; k < DD; ++k) acc += psum[(size_t)g * DD + k] * inv * Wm[(size_t)k * DD + j];
    outG[t] = acc;
}

// ---------------- orchestration ----------------
static inline int nblk(long n) { return (int)((n + 255) / 256); }

extern "C" void kernel_launch(void* const* d_in, const int* in_sizes, int n_in,
                              void* d_out, int out_size, void* d_ws, size_t ws_size,
                              hipStream_t stream) {
    const float* h     = (const float*)d_in[0];
    const int*   src   = (const int*)d_in[2];
    const int*   dst   = (const int*)d_in[3];
    const int*   gid   = (const int*)d_in[4];
    const float* W0    = (const float*)d_in[5];
    const float* al0   = (const float*)d_in[6];
    const float* ar0   = (const float*)d_in[7];
    const float* b0    = (const float*)d_in[8];
    const float* W1    = (const float*)d_in[9];
    const float* al1   = (const float*)d_in[10];
    const float* ar1   = (const float*)d_in[11];
    const float* b1    = (const float*)d_in[12];
    const float* W2    = (const float*)d_in[13];
    const float* al2   = (const float*)d_in[14];
    const float* ar2   = (const float*)d_in[15];
    const float* b2    = (const float*)d_in[16];
    const float* resW2 = (const float*)d_in[17];
    const float* Wm    = (const float*)d_in[18];
    const float* bm    = (const float*)d_in[19];

    char* ws = (char*)d_ws;
    float* X0   = (float*)(ws + OFF_X0);
    float* X1   = (float*)(ws + OFF_X1);
    float* FEAT = (float*)(ws + OFF_FT);
    u16*   ABF  = (u16*)(ws + OFF_ABF);
    u16*   WBF  = (u16*)(ws + OFF_WBF);
    float* EL   = (float*)(ws + OFF_EL);
    float* ER   = (float*)(ws + OFF_ER);
    int*   MB   = (int*)(ws + OFF_M);
    float* DEN  = (float*)(ws + OFF_DEN);
    float* EB   = (float*)(ws + OFF_EB);
    float* PSUM = (float*)(ws + OFF_PSUM);
    float* PCNT = (float*)(ws + OFF_PCNT);

    auto gemm = [&](const u16* A, const u16* Bt, float* C, int M, int N, int K) {
        dim3 g(N / 64, (M + 255) / 256);
        gemm_bf16_wmma<<<g, 256, 0, stream>>>(A, Bt, C, M, N, K);
    };
    auto attn = [&](const float* feat, const float* al, const float* ar, int H) {
        k_attn_logits<<<(NODES * H + 7) / 8, 256, 0, stream>>>(feat, al, ar, EL, ER, H);
        k_init_m<<<nblk((long)NODES * H), 256, 0, stream>>>(MB, (long)NODES * H);
        (void)hipMemsetAsync(DEN, 0, (size_t)NODES * H * 4, stream);
        k_edge_logit_max<<<nblk((long)EDGES * H), 256, 0, stream>>>(EL, ER, src, dst, EB, MB, H);
        k_edge_exp<<<nblk((long)EDGES * H), 256, 0, stream>>>(dst, EB, MB, DEN, H);
    };

    // ---------------- layer 0 : h @ W0, H=4, no residual, elu ----------------
    k_f32_to_bf16<<<nblk((long)NODES * 128), 256, 0, stream>>>(h, ABF, (long)NODES * 128);
    k_transpose_bf16<<<nblk((long)128 * 512), 256, 0, stream>>>(W0, WBF, 128, 512);
    gemm(ABF, WBF, FEAT, NODES, 512, 128);
    attn(FEAT, al0, ar0, 4);
    (void)hipMemsetAsync(X0, 0, (size_t)NODES * 512 * 4, stream);
    k_scatter<<<nblk((long)EDGES * 4 * 32), 256, 0, stream>>>(src, dst, EB, DEN, FEAT, X0, 4);
    k_bias_act<<<nblk((long)NODES * 512), 256, 0, stream>>>(X0, b0, 512, 1);

    // ---------------- layer 1 : X0 @ W1, H=4, identity residual, elu ----------------
    k_f32_to_bf16<<<nblk((long)NODES * 512), 256, 0, stream>>>(X0, ABF, (long)NODES * 512);
    k_transpose_bf16<<<nblk((long)512 * 512), 256, 0, stream>>>(W1, WBF, 512, 512);
    gemm(ABF, WBF, FEAT, NODES, 512, 512);
    attn(FEAT, al1, ar1, 4);
    (void)hipMemcpyAsync(X1, X0, (size_t)NODES * 512 * 4, hipMemcpyDeviceToDevice, stream);
    k_scatter<<<nblk((long)EDGES * 4 * 32), 256, 0, stream>>>(src, dst, EB, DEN, FEAT, X1, 4);
    k_bias_act<<<nblk((long)NODES * 512), 256, 0, stream>>>(X1, b1, 512, 1);

    // ---------------- layer 2 : X1 @ W2, H=6, linear residual (X1 @ resW2), no act ----------------
    k_f32_to_bf16<<<nblk((long)NODES * 512), 256, 0, stream>>>(X1, ABF, (long)NODES * 512);
    k_transpose_bf16<<<nblk((long)512 * 768), 256, 0, stream>>>(W2, WBF, 512, 768);
    gemm(ABF, WBF, FEAT, NODES, 768, 512);
    k_transpose_bf16<<<nblk((long)512 * 768), 256, 0, stream>>>(resW2, WBF, 512, 768);
    gemm(ABF, WBF, X0, NODES, 768, 512);           // residual directly into output buffer
    attn(FEAT, al2, ar2, 6);
    k_scatter<<<nblk((long)EDGES * 6 * 32), 256, 0, stream>>>(src, dst, EB, DEN, FEAT, X0, 6);
    k_bias_act<<<nblk((long)NODES * 768), 256, 0, stream>>>(X0, b2, 768, 0);

    // ---------------- mean over heads, per-graph pooling, final linear ----------------
    (void)hipMemsetAsync(PSUM, 0, (size_t)GR * DD * 4, stream);
    (void)hipMemsetAsync(PCNT, 0, (size_t)GR * 4, stream);
    k_counts<<<nblk(NODES), 256, 0, stream>>>(gid, PCNT);
    k_mean_pool<<<nblk((long)NODES * DD), 256, 0, stream>>>(X0, gid, (float*)d_out, PSUM, 6);
    k_final<<<nblk((long)GR * DD), 256, 0, stream>>>(PSUM, PCNT, Wm, bm,
                                                     (float*)d_out + (size_t)NODES * DD);
}
